// FocalSparseBEVBackBone_37958920962322
// MI455X (gfx1250) — compile-verified, matrix operand order checked by reading, not run on hardware
//
#include <hip/hip_runtime.h>
#include <hip/hip_bf16.h>

// ---------------------------------------------------------------------------
// CDNA5 / gfx1250 WMMA + TDM types
// ---------------------------------------------------------------------------
typedef __attribute__((ext_vector_type(16))) _Float16 v16h;
typedef __attribute__((ext_vector_type(8)))  _Float16 v8h;
typedef __attribute__((ext_vector_type(8)))  float    v8f;
typedef __attribute__((ext_vector_type(4)))  unsigned int v4u;
typedef __attribute__((ext_vector_type(8)))  int      v8i;
typedef __attribute__((ext_vector_type(4)))  int      v4i;

union AFrag { v16h v; _Float16 h[16]; v8h half8[2]; };
union CFrag { v8f  v; float    f[8];  };

#define LDS_PAD_H 8   // +8 halves (16B / 4 dwords) per row: de-conflicts LDS banks

// ---------------------------------------------------------------------------
// Sparse conv tile block: 4 waves x 16-row tiles, one shared 16-col B tile.
// B tile (16 x Cs f16, row-skewed by 16B) is staged into LDS by the Tensor
// Data Mover (TENSORcnt), then each wave streams A rows with global b128 loads
// and reads B fragments with ds_load_b128.
//   a16: (Nin, Cs) f16, rows zero-padded to Cs (mult. of 32)
//   wt : (Cout, Cs) f16 for ONE kernel offset (transposed + padded)
// A fragment (ISA 7.12.2): lane holds row mr, K in {k0..k0+7} U {k0+16..k0+23},
// k0 = 8*(lane>>4) -> two contiguous 16B runs.
// ---------------------------------------------------------------------------
__global__ __launch_bounds__(128)
void spconv_tile_wmma(const _Float16* __restrict__ a16,
                      const _Float16* __restrict__ wt,
                      const int* __restrict__ ii, const int* __restrict__ oi,
                      int L, int Cs, int Cout, float* __restrict__ out)
{
    __shared__ _Float16 sB[16 * (256 + LDS_PAD_H)];   // worst case Cs = 256
    __shared__ int s_ii[4][16], s_oi[4][16];

    const int tid  = threadIdx.x;
    const int wave = tid >> 5;
    const int lane = tid & 31;
    const int half = lane >> 4;
    const int mr   = lane & 15;
    const int rbase = (blockIdx.x * 4 + wave) * 16;
    const int cbase = blockIdx.y * 16;
    const int lstr  = Cs + LDS_PAD_H;

    // ---- wave 0: TDM-load the 16 x Cs B tile (rows n = cbase..cbase+15) ----
    if (tid < 32) {
        unsigned lds_off = (unsigned)(uintptr_t)(&sB[0]);
        unsigned long long ga =
            (unsigned long long)(uintptr_t)(wt + (size_t)cbase * Cs);
        int rows = Cout - cbase;                       // >= 16
        // pad_interval: 1<<iv qwords per row = Cs/4  ->  iv = log2(Cs) - 2
        unsigned iv = (unsigned)((31 - __builtin_clz((unsigned)Cs)) - 2);

        v4u g0; v8i g1; v4i g2; v4i g3; v8i g4;
        g0[0] = 1u;                                    // count=1, user mode
        g0[1] = lds_off;                               // lds_addr
        g0[2] = (unsigned)(ga & 0xffffffffu);          // global_addr[31:0]
        g0[3] = (unsigned)((ga >> 32) & 0x01ffffffu)   // global_addr[56:32]
              | (2u << 30);                            // type = 2 ("image")
        g1[0] = (int)((1u << 16)                       // data_size = 1 (2B)
              | (1u << 20)                             // pad_enable
              | (iv << 22)                             // pad_interval
              | (3u << 25));                           // pad_amount = 4 dwords (16B)
        g1[1] = (int)(((unsigned)Cs & 0xffffu) << 16); // tensor_dim0[15:0]
        g1[2] = (int)((((unsigned)rows & 0xffffu) << 16)); // dim0[31:16]=0 | tensor_dim1 lo
        g1[3] = (int)(((unsigned)Cs & 0xffffu) << 16); // tensor_dim1 hi=0 | tile_dim0 = Cs
        g1[4] = 16;                                    // tile_dim1 = 16, tile_dim2 = 0
        g1[5] = Cs;                                    // tensor_dim0_stride
        g1[6] = 0; g1[7] = 0;                          // dim1_stride unused (2D)
        g2 = (v4i)(0); g3 = (v4i)(0); g4 = (v8i)(0);
        __builtin_amdgcn_tensor_load_to_lds(g0, g1, g2, g3, g4, 0);
        __builtin_amdgcn_s_wait_tensorcnt(0);
    }
    if (lane < 16) {
        int r = rbase + lane;
        s_ii[wave][lane] = (r < L) ? ii[r] : 0;   // clamped rows compute garbage...
        s_oi[wave][lane] = (r < L) ? oi[r] : -1;  // ...but are never scattered
    }
    __syncthreads();

    const _Float16* arow  = a16 + (size_t)s_ii[wave][mr] * Cs + half * 8;
    const _Float16* sBrow = sB + (size_t)mr * lstr + half * 8;

    v8f acc = {};
    for (int kb = 0; kb < Cs; kb += 32) {
        AFrag a, b;
        a.half8[0] = *(const v8h*)(arow + kb);
        a.half8[1] = *(const v8h*)(arow + kb + 16);
        b.half8[0] = *(const v8h*)(sBrow + kb);
        b.half8[1] = *(const v8h*)(sBrow + kb + 16);
        if (kb + 32 < Cs)
            __builtin_prefetch(arow + kb + 32, 0, 3);  // global_prefetch_b8
        acc = __builtin_amdgcn_wmma_f32_16x16x32_f16(false, a.v, false, b.v,
                                                     (short)0, acc, false, false);
    }

    CFrag cf; cf.v = acc;
    const int mycol = cbase + mr;
#pragma unroll
    for (int r = 0; r < 8; ++r) {
        int orow = s_oi[wave][r + 8 * half];
        if (orow >= 0)
            atomicAdd(&out[(size_t)orow * Cout + mycol], cf.f[r]);
    }
}

// ---------------------------------------------------------------------------
// Dense 3x3 conv, implicit GEMM with WMMA.
// xpad: (B, H+2, W+2, Cin) f16 NHWC, spatially zero-padded
// wt  : (Cout, 9*Cin) f16 with K ordered k = tap*Cin + cin
// y   : (B, Cout, H, W) f32
// ---------------------------------------------------------------------------
__global__ __launch_bounds__(32)
void conv3x3_wmma(const _Float16* __restrict__ xpad,
                  const _Float16* __restrict__ wt,
                  const float* __restrict__ bias, float* __restrict__ y,
                  int B, int Cin, int Cout, int H, int W)
{
    const int lane = threadIdx.x;
    const int half = lane >> 4;
    const int mr   = lane & 15;
    const int pbase = blockIdx.x * 16;
    const int cbase = blockIdx.y * 16;
    const int b     = blockIdx.z;
    const int HW = H * W, Wp = W + 2, Hp = H + 2;
    const int K = 9 * Cin;

    const int p  = pbase + mr;          // HW % 16 == 0 -> always valid
    const int py = p / W, px = p % W;
    const _Float16* xb   = xpad + (size_t)b * Hp * Wp * Cin;
    const _Float16* brow = wt + (size_t)(cbase + mr) * K + half * 8;
    const int n = cbase + mr;

    v8f acc = {};
    for (int kb = 0; kb < K; kb += 32) {
        AFrag a, bf;
        bf.half8[0] = *(const v8h*)(brow + kb);
        bf.half8[1] = *(const v8h*)(brow + kb + 16);
#pragma unroll
        for (int r = 0; r < 2; ++r) {
            int k   = kb + half * 8 + r * 16;
            int tap = k / Cin;
            int cin = k - tap * Cin;
            int yy = py + tap / 3, xx = px + tap % 3;   // pad absorbs the -1
            a.half8[r] = *(const v8h*)(xb + ((size_t)yy * Wp + xx) * Cin + cin);
        }
        if (kb + 32 < K)
            __builtin_prefetch(brow + kb + 32, 0, 3);
        acc = __builtin_amdgcn_wmma_f32_16x16x32_f16(false, a.v, false, bf.v,
                                                     (short)0, acc, false, false);
    }

    CFrag cf; cf.v = acc;
    const float bia = bias ? bias[n] : 0.0f;
#pragma unroll
    for (int r = 0; r < 8; ++r) {
        int pp = pbase + r + 8 * half;
        y[((size_t)b * Cout + n) * HW + pp] = cf.f[r] + bia;
    }
}

// ---------------------------------------------------------------------------
// Layout-conversion kernels (elementwise, bandwidth-trivial vs the GEMMs)
// ---------------------------------------------------------------------------
__global__ void cvt_pad_f16_k(const float* __restrict__ src, _Float16* __restrict__ dst,
                              long long nrows, int Cin, int Cs)
{
    long long i = (long long)blockIdx.x * blockDim.x + threadIdx.x;
    if (i < nrows * (long long)Cs) {
        long long r = i / Cs; int c = (int)(i % Cs);
        dst[i] = (c < Cin) ? (_Float16)src[r * Cin + c] : (_Float16)0.0f;
    }
}

// Wt[k_off][n][kk(pad Cs)] = W[k_off][kk][n]   (W: (K, Cin, Cout) f32)
__global__ void wtrans_pad_f16_k(const float* __restrict__ W, _Float16* __restrict__ Wt,
                                 int K, int Cin, int Cout, int Cs)
{
    long long i = (long long)blockIdx.x * blockDim.x + threadIdx.x;
    long long tot = (long long)K * Cout * Cs;
    if (i < tot) {
        int ko  = (int)(i / ((long long)Cout * Cs));
        long long rem = i % ((long long)Cout * Cs);
        int n  = (int)(rem / Cs);
        int kk = (int)(rem % Cs);
        Wt[i] = (kk < Cin) ? (_Float16)W[((size_t)ko * Cin + kk) * Cout + n]
                           : (_Float16)0.0f;
    }
}

// Wt[n][tap*Cin + cin] = (f16)W[((n*Cin)+cin)*9 + tap]   (W: OIHW)
__global__ void wtrans3_f16_k(const float* __restrict__ W, _Float16* __restrict__ Wt,
                              int Cout, int Cin)
{
    long long i = (long long)blockIdx.x * blockDim.x + threadIdx.x;
    long long tot = (long long)Cout * Cin * 9;
    if (i < tot) {
        int n = (int)(i / ((long long)Cin * 9));
        long long rem = i % ((long long)Cin * 9);
        int tap = (int)(rem / Cin);
        int cin = (int)(rem % Cin);
        Wt[i] = (_Float16)W[((size_t)n * Cin + cin) * 9 + tap];
    }
}

// NCHW f32 -> NHWC f16 with 1-pixel zero pad: dst (B, H+2, W+2, C)
__global__ void pad_nhwc_f16_k(const float* __restrict__ x, _Float16* __restrict__ dst,
                               int B, int C, int H, int W)
{
    const int Hp = H + 2, Wp = W + 2;
    long long i = (long long)blockIdx.x * blockDim.x + threadIdx.x;
    long long tot = (long long)B * Hp * Wp * C;
    if (i < tot) {
        int c  = (int)(i % C);
        long long r = i / C;
        int xx = (int)(r % Wp); r /= Wp;
        int yy = (int)(r % Hp);
        int b  = (int)(r / Hp);
        float v = 0.0f;
        int sy = yy - 1, sx = xx - 1;
        if (sy >= 0 && sy < H && sx >= 0 && sx < W)
            v = x[(((size_t)b * C + c) * H + sy) * W + sx];
        dst[i] = (_Float16)v;
    }
}

// ---------------------------------------------------------------------------
// Elementwise / utility kernels
// ---------------------------------------------------------------------------
__global__ void fill_bias_k(float* out, const float* bias, long long n, int C)
{
    long long i = (long long)blockIdx.x * blockDim.x + threadIdx.x;
    if (i < n * (long long)C) out[i] = bias ? bias[(int)(i % C)] : 0.0f;
}

__global__ void zero_k(float* p, long long n)
{
    long long i = (long long)blockIdx.x * blockDim.x + threadIdx.x;
    if (i < n) p[i] = 0.0f;
}

__global__ void bn_act_k(const float* __restrict__ x,
                         const float* __restrict__ g, const float* __restrict__ bb,
                         const float* __restrict__ mu, const float* __restrict__ vr,
                         const float* __restrict__ res, float* __restrict__ y,
                         long long n, int C, float eps, int relu)
{
    long long i = (long long)blockIdx.x * blockDim.x + threadIdx.x;
    if (i < n * (long long)C) {
        int c = (int)(i % C);
        float v = (x[i] - mu[c]) * (g[c] * rsqrtf(vr[c] + eps)) + bb[c];
        if (res) v += res[i];
        if (relu) v = fmaxf(v, 0.0f);
        y[i] = v;
    }
}

__global__ void bn4_relu_k(const float* __restrict__ x,
                           const float* __restrict__ g, const float* __restrict__ bb,
                           const float* __restrict__ mu, const float* __restrict__ vr,
                           float* __restrict__ y, int C, long long HW,
                           long long total, float eps)
{
    long long i = (long long)blockIdx.x * blockDim.x + threadIdx.x;
    if (i < total) {
        int c = (int)((i / HW) % C);
        float v = (x[i] - mu[c]) * (g[c] * rsqrtf(vr[c] + eps)) + bb[c];
        y[i] = fmaxf(v, 0.0f);
    }
}

__global__ void scatter_add_rows_k(const float* __restrict__ src,
                                   const int* __restrict__ map,
                                   float* __restrict__ dst, long long n, int C)
{
    long long i = (long long)blockIdx.x * blockDim.x + threadIdx.x;
    if (i < n * (long long)C) {
        long long r = i / C; int c = (int)(i % C);
        atomicAdd(&dst[(size_t)map[r] * C + c], src[i]);
    }
}

__global__ void scatter_dense_k(const float* __restrict__ src,
                                const int* __restrict__ flat,
                                float* __restrict__ dst, long long n, int C, long long HW)
{
    long long i = (long long)blockIdx.x * blockDim.x + threadIdx.x;
    if (i < n * (long long)C) {
        long long r = i / C; int c = (int)(i % C);
        long long fp = flat[r];
        long long b = fp / HW, p = fp % HW;
        dst[(b * C + c) * HW + p] = src[i];
    }
}

__global__ __launch_bounds__(256)
void gn_gelu_k(const float* __restrict__ x, const float* __restrict__ gg,
               const float* __restrict__ gb, float* __restrict__ y,
               int C, long long HW, int groups)
{
    int b = blockIdx.x / groups, g = blockIdx.x % groups;
    int cpg = C / groups;
    long long n = (long long)cpg * HW;
    const float* base = x + ((size_t)b * C + (size_t)g * cpg) * HW;
    float*       yb   = y + ((size_t)b * C + (size_t)g * cpg) * HW;

    float s = 0.f, s2 = 0.f;
    for (long long i = threadIdx.x; i < n; i += blockDim.x) {
        float v = base[i]; s += v; s2 += v * v;
    }
    __shared__ float rs[256], rq[256];
    rs[threadIdx.x] = s; rq[threadIdx.x] = s2;
    __syncthreads();
    for (int o = 128; o > 0; o >>= 1) {
        if (threadIdx.x < (unsigned)o) {
            rs[threadIdx.x] += rs[threadIdx.x + o];
            rq[threadIdx.x] += rq[threadIdx.x + o];
        }
        __syncthreads();
    }
    float mean = rs[0] / (float)n;
    float var  = rq[0] / (float)n - mean * mean;
    float inv  = rsqrtf(var + 1e-5f);
    for (long long i = threadIdx.x; i < n; i += blockDim.x) {
        int c = g * cpg + (int)(i / HW);
        float v = (base[i] - mean) * inv * gg[c] + gb[c];
        v = 0.5f * v * (1.0f + erff(v * 0.70710678118f));
        yb[i] = v;
    }
}

__global__ void head_score_k(const float* __restrict__ s, const float* __restrict__ Wh,
                             const float* __restrict__ bh, float* __restrict__ score,
                             int B, int C, long long HW, int ncls)
{
    long long idx = (long long)blockIdx.x * blockDim.x + threadIdx.x;
    if (idx < (long long)B * HW) {
        int b = (int)(idx / HW); long long p = idx % HW;
        const float* xb = s + (size_t)b * C * HW + p;
        float mx = -3.4e38f;
        for (int k = 0; k < ncls; ++k) {
            float acc = bh[k];
            for (int c = 0; c < C; ++c) acc += xb[(size_t)c * HW] * Wh[k * C + c];
            mx = fmaxf(mx, acc);
        }
        score[idx] = mx;
    }
}

__global__ void rank_scatter_k(const float* __restrict__ score, int* __restrict__ order,
                               int B, int HW)
{
    long long idx = (long long)blockIdx.x * blockDim.x + threadIdx.x;
    if (idx < (long long)B * HW) {
        int b = (int)(idx / HW); int i = (int)(idx % HW);
        const float* sb = score + (size_t)b * HW;
        float si = sb[i];
        int r = 0;
        for (int j = 0; j < HW; ++j) {
            float sj = sb[j];
            r += (sj > si) || (sj == si && j < i);
        }
        order[(size_t)b * HW + r] = i;
    }
}

__global__ void write_out_k(const float* __restrict__ pillar, const int* __restrict__ order,
                            const float* __restrict__ score, float* __restrict__ out,
                            int B, int C, long long HW, int ns, long long total)
{
    long long i = (long long)blockIdx.x * blockDim.x + threadIdx.x;
    if (i >= total) return;
    long long selN = (long long)B * ns * C;
    if (i < selN) {
        int b = (int)(i / ((long long)ns * C));
        long long rem = i % ((long long)ns * C);
        int r = (int)(rem / C), c = (int)(rem % C);
        int p = order[(size_t)b * HW + r];
        out[i] = pillar[((size_t)b * C + c) * HW + p];
    } else if (i < selN + (long long)B * HW) {
        out[i] = score[i - selN];
    } else {
        out[i] = (float)order[i - selN - (long long)B * HW];
    }
}

// ---------------------------------------------------------------------------
// Host orchestration
// ---------------------------------------------------------------------------
struct BNp  { const float *g, *b, *m, *v; };
struct Blk  { const float *W1, *b1; BNp bn1; const float *W2, *b2; BNp bn2; };
struct Pair { const int *ii, *oi; int L; };

extern "C" void kernel_launch(void* const* d_in, const int* in_sizes, int n_in,
                              void* d_out, int out_size, void* d_ws, size_t ws_size,
                              hipStream_t stream)
{
    int cur = 0;
    auto F = [&]() -> const float* {
        int i = (cur < n_in) ? cur : (n_in - 1); ++cur; return (const float*)d_in[i];
    };
    auto P = [&]() -> Pair {
        Pair p; int i0 = (cur < n_in) ? cur : (n_in - 1);
        p.L = in_sizes[i0]; p.ii = (const int*)d_in[i0]; ++cur;
        int i1 = (cur < n_in) ? cur : (n_in - 1);
        p.oi = (const int*)d_in[i1]; ++cur;
        return p;
    };
    auto BNr = [&]() -> BNp { BNp b; b.g = F(); b.b = F(); b.m = F(); b.v = F(); return b; };
    auto BLK = [&]() -> Blk {
        Blk b; b.W1 = F(); b.b1 = F(); b.bn1 = BNr();
        b.W2 = F(); b.b2 = F(); b.bn2 = BNr(); return b;
    };

    // ---- inputs, in setup_inputs() depth-first dict order ----
    const float* vf = F();
    const int NV = in_sizes[0] / 5;

    const float* Win = F(); BNp bnin = BNr();
    Blk s1b[2]; s1b[0] = BLK(); s1b[1] = BLK();
    struct Stage { const float* Wd; BNp bnd; Blk blk[2]; } st[5];
    for (int i = 0; i < 5; ++i) { st[i].Wd = F(); st[i].bnd = BNr(); st[i].blk[0] = BLK(); st[i].blk[1] = BLK(); }
    const float* Wout = F(); BNp bnout = BNr();
    const float* Wsh  = F(); const float* bsh = F(); BNp bnsh = BNr();
    const float* Wmod = F(); BNp bnmod = BNr();
    const float* Wsem = F(); const float* bsem = F();
    const float* semgg = F(); const float* semgb = F();
    const float* Whead = F(); const float* bhead = F();

    Pair rb0[27];       for (int k = 0; k < 27; ++k) rb0[k] = P();
    Pair subm[5][27];   for (int i = 0; i < 5; ++i) for (int k = 0; k < 27; ++k) subm[i][k] = P();
    Pair down[5][27];   for (int i = 0; i < 5; ++i) for (int k = 0; k < 27; ++k) down[i][k] = P();
    cur += 5;                                   // nout scalars (recovered from rulebooks)
    const int* invp = (const int*)d_in[(cur < n_in) ? cur : (n_in - 1)]; ++cur;
    cur += 1;                                   // nbev scalar
    Pair pout[9];       for (int k = 0; k < 9; ++k) pout[k] = P();
    cur += 1;                                   // n2 scalar
    Pair rbsh[9];       for (int k = 0; k < 9; ++k) rbsh[k] = P();
    const int* flatp = (const int*)d_in[(cur < n_in) ? cur : (n_in - 1)]; ++cur;
    cur += 4;                                   // bs, H, Wd, ratio (static)

    int nout[5]; for (int i = 0; i < 5; ++i) nout[i] = subm[i][13].L;
    const int nbev = pout[4].L;
    const int n2   = rbsh[4].L;
    const int B = 2, H = 128, W = 128;
    const long long HW = (long long)H * W;
    const int CH[5] = {32, 64, 128, 128, 128};

    // ---- workspace bump allocator ----
    char* wsb = (char*)d_ws; size_t off = 0;
    auto alloc_b = [&](size_t bytes) -> char* {
        size_t a = (off + 255) & ~(size_t)255;
        off = a + bytes;
        return wsb + a;
    };
    auto alloc_f = [&](size_t n) -> float*     { return (float*)alloc_b(n * 4); };
    auto alloc_h = [&](size_t n) -> _Float16*  { return (_Float16*)alloc_b(n * 2); };

    float* f0 = alloc_f((size_t)NV * 16);
    float* fl[5]; for (int i = 0; i < 5; ++i) fl[i] = alloc_f((size_t)nout[i] * CH[i]);
    size_t maxRC = (size_t)NV * 16;
    for (int i = 0; i < 5; ++i) if ((size_t)nout[i] * CH[i] > maxRC) maxRC = (size_t)nout[i] * CH[i];
    if ((size_t)n2 * 256 > maxRC) maxRC = (size_t)n2 * 256;
    float* tmpA = alloc_f(maxRC);
    float* tmpB = alloc_f(maxRC);
    float* bev  = alloc_f((size_t)nbev * 128);
    float* o1   = alloc_f((size_t)n2 * 256);
    float* o2   = alloc_f((size_t)n2 * 256);
    float* dense  = alloc_f((size_t)B * 256 * HW);   // reused as sem output
    float* pillar = alloc_f((size_t)B * 256 * HW);
    float* score  = alloc_f((size_t)B * HW);
    int*   order  = (int*)alloc_b((size_t)B * HW * 4);

    size_t maxF16 = (size_t)NV * 32;
    for (int i = 0; i < 5; ++i) {
        int Cs = (CH[i] + 31) & ~31;
        if ((size_t)nout[i] * Cs > maxF16) maxF16 = (size_t)nout[i] * Cs;
    }
    if ((size_t)nbev * 128 > maxF16) maxF16 = (size_t)nbev * 128;
    if ((size_t)n2 * 256 > maxF16)   maxF16 = (size_t)n2 * 256;
    _Float16* feat16 = alloc_h(maxF16);
    _Float16* wt16   = alloc_h((size_t)27 * 256 * 256);
    _Float16* xpad   = alloc_h((size_t)B * (H + 2) * (W + 2) * 256);
    _Float16* wt3    = alloc_h((size_t)256 * 9 * 256);
    (void)ws_size;

    auto g1 = [](long long n, int t) { return dim3((unsigned)((n + t - 1) / t)); };

    // gather -> f16 WMMA GEMM per offset -> scatter-add
    auto run_spc = [&](const float* fin, const float* Wall, const Pair* pairs, int K,
                       long long nrows_in, int nrows_out, int Cin, int Cout,
                       const float* bias, float* out) {
        const int Cs = (Cin + 31) & ~31;
        cvt_pad_f16_k<<<g1(nrows_in * Cs, 256), 256, 0, stream>>>(fin, feat16, nrows_in, Cin, Cs);
        wtrans_pad_f16_k<<<g1((long long)K * Cout * Cs, 256), 256, 0, stream>>>(Wall, wt16, K, Cin, Cout, Cs);
        fill_bias_k<<<g1((long long)nrows_out * Cout, 256), 256, 0, stream>>>(out, bias, nrows_out, Cout);
        for (int k = 0; k < K; ++k) {
            if (pairs[k].L <= 0) continue;
            dim3 grid((pairs[k].L + 63) / 64, Cout / 16);   // 4 row-tiles per block
            spconv_tile_wmma<<<grid, 128, 0, stream>>>(
                feat16, wt16 + (size_t)k * Cout * Cs,
                pairs[k].ii, pairs[k].oi, pairs[k].L, Cs, Cout, out);
        }
    };
    auto run_bn = [&](const float* x, const BNp& bn, const float* res, float* y,
                      long long n, int C, float eps, int relu) {
        bn_act_k<<<g1(n * C, 256), 256, 0, stream>>>(x, bn.g, bn.b, bn.m, bn.v,
                                                     res, y, n, C, eps, relu);
    };
    auto run_block = [&](float* f, const Blk& b, const Pair* pairs, int K, int n, int C) {
        run_spc(f, b.W1, pairs, K, n, n, C, C, b.b1, tmpA);
        run_bn(tmpA, b.bn1, nullptr, tmpA, n, C, 1e-3f, 1);
        run_spc(tmpA, b.W2, pairs, K, n, n, C, C, b.b2, tmpB);
        run_bn(tmpB, b.bn2, f, f, n, C, 1e-3f, 1);       // relu(bn(h) + f) -> f
    };

    // ---- sparse backbone ----
    run_spc(vf, Win, rb0, 27, NV, NV, 5, 16, nullptr, f0);
    run_bn(f0, bnin, nullptr, f0, NV, 16, 1e-3f, 1);
    run_block(f0, s1b[0], rb0, 27, NV, 16);
    run_block(f0, s1b[1], rb0, 27, NV, 16);

    const float* fin = f0; long long nin = NV; int cin = 16;
    for (int i = 0; i < 5; ++i) {
        run_spc(fin, st[i].Wd, down[i], 27, nin, nout[i], cin, CH[i], nullptr, fl[i]);
        run_bn(fl[i], st[i].bnd, nullptr, fl[i], nout[i], CH[i], 1e-3f, 1);
        run_block(fl[i], st[i].blk[0], subm[i], 27, nout[i], CH[i]);
        run_block(fl[i], st[i].blk[1], subm[i], 27, nout[i], CH[i]);
        fin = fl[i]; nin = nout[i]; cin = CH[i];
    }

    // ---- BEV pool: bev[inv[r]] += cat[r] ----
    zero_k<<<g1((long long)nbev * 128, 256), 256, 0, stream>>>(bev, (long long)nbev * 128);
    scatter_add_rows_k<<<g1((long long)nout[2] * 128, 256), 256, 0, stream>>>(fl[2], invp, bev, nout[2], 128);
    scatter_add_rows_k<<<g1((long long)nout[3] * 128, 256), 256, 0, stream>>>(fl[3], invp + nout[2], bev, nout[3], 128);
    scatter_add_rows_k<<<g1((long long)nout[4] * 128, 256), 256, 0, stream>>>(fl[4], invp + nout[2] + nout[3], bev, nout[4], 128);

    // ---- BEV sparse convs ----
    run_spc(bev, Wout, pout, 9, nbev, n2, 128, 256, nullptr, o1);
    run_bn(o1, bnout, nullptr, o1, n2, 256, 1e-3f, 1);
    run_spc(o1, Wsh, rbsh, 9, n2, n2, 256, 256, bsh, o2);
    run_bn(o2, bnsh, nullptr, o2, n2, 256, 1e-3f, 1);

    // ---- densify + dense convs (WMMA implicit GEMM, NHWC f16 padded input) ----
    zero_k<<<g1((long long)B * 256 * HW, 256), 256, 0, stream>>>(dense, (long long)B * 256 * HW);
    scatter_dense_k<<<g1((long long)n2 * 256, 256), 256, 0, stream>>>(o2, flatp, dense, n2, 256, HW);

    {
        const long long padN = (long long)B * (H + 2) * (W + 2) * 256;
        dim3 grid((unsigned)(HW / 16), 256 / 16, B);

        pad_nhwc_f16_k<<<g1(padN, 256), 256, 0, stream>>>(dense, xpad, B, 256, H, W);
        wtrans3_f16_k<<<g1((long long)256 * 256 * 9, 256), 256, 0, stream>>>(Wmod, wt3, 256, 256);
        conv3x3_wmma<<<grid, 32, 0, stream>>>(xpad, wt3, nullptr, pillar, B, 256, 256, H, W);
        bn4_relu_k<<<g1((long long)B * 256 * HW, 256), 256, 0, stream>>>(
            pillar, bnmod.g, bnmod.b, bnmod.m, bnmod.v, pillar, 256, HW, (long long)B * 256 * HW, 1e-5f);

        pad_nhwc_f16_k<<<g1(padN, 256), 256, 0, stream>>>(pillar, xpad, B, 256, H, W);
        wtrans3_f16_k<<<g1((long long)256 * 256 * 9, 256), 256, 0, stream>>>(Wsem, wt3, 256, 256);
        conv3x3_wmma<<<grid, 32, 0, stream>>>(xpad, wt3, bsem, dense, B, 256, 256, H, W);
        gn_gelu_k<<<dim3(B * 32), 256, 0, stream>>>(dense, semgg, semgb, dense, 256, HW, 32);
    }

    // ---- head + score + stable argsort + output packing ----
    head_score_k<<<g1((long long)B * HW, 256), 256, 0, stream>>>(dense, Whead, bhead, score, B, 256, HW, 10);
    rank_scatter_k<<<g1((long long)B * HW, 256), 256, 0, stream>>>(score, order, B, (int)HW);

    const int ns = (int)(HW / 2);   // ratio = 0.5
    write_out_k<<<g1((long long)out_size, 256), 256, 0, stream>>>(
        pillar, order, score, (float*)d_out, B, 256, HW, ns, (long long)out_size);
}
